// MultiHeadAttention_72404558675970
// MI455X (gfx1250) — compile-verified, matrix operand order checked by reading, not run on hardware
//
#include <hip/hip_runtime.h>
#include <hip/hip_bf16.h>

// ---- problem constants ----
constexpr int B_ = 4;
constexpr int T_ = 1024;
constexpr int C_ = 1024;
constexpr int H_ = 16;
constexpr int D_ = 64;
constexpr int M_ = B_ * T_;   // 4096 rows

typedef _Float16 half16 __attribute__((ext_vector_type(16)));
typedef _Float16 half8  __attribute__((ext_vector_type(8)));
typedef float    float8 __attribute__((ext_vector_type(8)));
typedef int      int4v  __attribute__((ext_vector_type(4)));

#if defined(__has_builtin)
#if __has_builtin(__builtin_amdgcn_global_load_async_to_lds_b128)
#define ASYNC_LDS 1
#endif
#endif

// 16-byte global -> LDS copy (async on CDNA5, VGPR round-trip fallback)
__device__ __forceinline__ void cp16(const void* g, void* l) {
#ifdef ASYNC_LDS
    __builtin_amdgcn_global_load_async_to_lds_b128(
        (__attribute__((address_space(1))) int4v*)g,
        (__attribute__((address_space(3))) int4v*)l, 0, 0);
#else
    *(uint4*)l = *(const uint4*)g;
#endif
}

__device__ __forceinline__ void async_wait() {
#ifdef ASYNC_LDS
#if __has_builtin(__builtin_amdgcn_s_wait_asynccnt)
    __builtin_amdgcn_s_wait_asynccnt(0);
#else
    asm volatile("s_wait_asynccnt 0x0" ::: "memory");
#endif
#endif
}

// lane-xor butterfly step via ds_swizzle (imm pattern, no address VGPR)
template <int XM>
__device__ __forceinline__ float swz(float v) {
#if __has_builtin(__builtin_amdgcn_ds_swizzle)
    return __int_as_float(__builtin_amdgcn_ds_swizzle(__float_as_int(v), (XM << 10) | 0x1f));
#else
    return __shfl_xor(v, XM);
#endif
}

// WMMA f32 = f16(16x32) x f16(32x16) + f32
__device__ __forceinline__ float8 wmma_f16(half16 a, half16 b, float8 c) {
    return __builtin_amdgcn_wmma_f32_16x16x32_f16(false, a, false, b, (short)0, c, false, false);
}

// Load a 16x32 f16 fragment (A-layout; B uses same layout on a transposed tile).
// ISA 7.12.2: lane<16 holds row=lane, K = hi*8..+7 (v0..3) and +16 (v4..7).
__device__ __forceinline__ half16 ld_frag(const _Float16* tile, int row0, int stride, int koff) {
    int lane = threadIdx.x & 31;
    int r = lane & 15, hi = lane >> 4;
    const _Float16* p = tile + (size_t)(row0 + r) * stride + koff + hi * 8;
    union { half16 v; half8 h[2]; } u;
    u.h[0] = *(const half8*)(p);
    u.h[1] = *(const half8*)(p + 16);
    return u.v;
}

// ---------------- f32 -> f16 convert ----------------
__global__ __launch_bounds__(256) void conv_f16_kernel(const float* __restrict__ src,
                                                       _Float16* __restrict__ dst, int n) {
    int i = (blockIdx.x * 256 + threadIdx.x) * 4;
    if (i < n) {
        float4 v = *(const float4*)(src + i);
        dst[i + 0] = (_Float16)v.x;
        dst[i + 1] = (_Float16)v.y;
        dst[i + 2] = (_Float16)v.z;
        dst[i + 3] = (_Float16)v.w;
    }
}

// ---------------- W[K][N] f32 -> Wt[N][K] f16 (LDS tiled transpose) ----------------
__global__ __launch_bounds__(256) void transpose_f16_kernel(const float* __restrict__ W,
                                                            _Float16* __restrict__ Wt) {
    __shared__ float tile[64][65];
    int n0 = blockIdx.x * 64, k0 = blockIdx.y * 64;
    int j = threadIdx.x & 63;
    int i0 = threadIdx.x >> 6;
#pragma unroll
    for (int ii = 0; ii < 16; ii++) {
        int i = i0 + ii * 4;
        tile[i][j] = W[(size_t)(k0 + i) * C_ + n0 + j];
    }
    __syncthreads();
#pragma unroll
    for (int ii = 0; ii < 16; ii++) {
        int i = i0 + ii * 4;   // n index inside tile
        Wt[(size_t)(n0 + i) * C_ + k0 + j] = (_Float16)tile[j][i];
    }
}

// ---------------- WMMA GEMM: Out = A[M x K] * Bt^T + bias ----------------
// A f16 row-major [M][K], Bt f16 [N][K]. K-stage 64, double-buffered async LDS,
// one barrier per stage.
// MODE 0: f16 out -> [B,H,T,D]   (Q, K projections)
// MODE 1: f16 out -> [B,H,D,T]   (V projection, pre-transposed for PV)
// MODE 2: f32 out -> [M][N]      (output projection)
template <int MODE>
__global__ __launch_bounds__(256) void gemm_wmma_kernel(const _Float16* __restrict__ A,
                                                        const _Float16* __restrict__ Bt,
                                                        const float* __restrict__ bias,
                                                        void* __restrict__ Out) {
    constexpr int K = C_;
    constexpr int LSTR = 72;                 // halves: 144B row pitch, 16B aligned, conflict-free
    __shared__ _Float16 lA[2][128 * LSTR];
    __shared__ _Float16 lB[2][128 * LSTR];

    const int tid  = threadIdx.x;
    const int lane = tid & 31;
    const int wv   = tid >> 5;
    const int wm   = wv & 3;                 // 4 waves along M
    const int wn   = wv >> 2;                // 2 waves along N
    const int m0   = blockIdx.y * 128;
    const int n0   = blockIdx.x * 128;
    const int r16  = lane & 15, hi = lane >> 4;

    float8 acc[2][4];
#pragma unroll
    for (int mi = 0; mi < 2; mi++)
#pragma unroll
        for (int nf = 0; nf < 4; nf++)
#pragma unroll
            for (int i = 0; i < 8; i++) acc[mi][nf][i] = 0.0f;

    const int srow  = tid >> 1;              // 0..127
    const int sbase = (tid & 1) * 32;        // 32 halves (64B) per thread

    auto issue = [&](int kk, int buf) {
        const _Float16* ga = A + (size_t)(m0 + srow) * K + kk + sbase;
        _Float16* da = &lA[buf][srow * LSTR + sbase];
        cp16(ga, da); cp16(ga + 8, da + 8); cp16(ga + 16, da + 16); cp16(ga + 24, da + 24);
        const _Float16* gb = Bt + (size_t)(n0 + srow) * K + kk + sbase;
        _Float16* db = &lB[buf][srow * LSTR + sbase];
        cp16(gb, db); cp16(gb + 8, db + 8); cp16(gb + 16, db + 16); cp16(gb + 24, db + 24);
    };

    issue(0, 0);
    int buf = 0;
    for (int kk = 0; kk < K; kk += 64, buf ^= 1) {
        async_wait();
        __syncthreads();                     // codegen waits DS cnt before barrier -> WAR safe
        if (kk + 64 < K) issue(kk + 64, buf ^ 1);
        if (kk + 128 < K) {                  // prefetch one more stage ahead into L2/WGP$
            __builtin_prefetch(A + (size_t)(m0 + srow) * K + kk + 128 + sbase, 0, 1);
            __builtin_prefetch(Bt + (size_t)(n0 + srow) * K + kk + 128 + sbase, 0, 1);
        }

#pragma unroll
        for (int kh = 0; kh < 2; kh++) {     // two 32-wide K slices in this stage
            half16 af0 = ld_frag(lA[buf], wm * 32,      LSTR, kh * 32);
            half16 af1 = ld_frag(lA[buf], wm * 32 + 16, LSTR, kh * 32);
            half16 bf[4];
#pragma unroll
            for (int nf = 0; nf < 4; nf++)
                bf[nf] = ld_frag(lB[buf], wn * 64 + nf * 16, LSTR, kh * 32);
#pragma unroll
            for (int nf = 0; nf < 4; nf++) acc[0][nf] = wmma_f16(af0, bf[nf], acc[0][nf]);
#pragma unroll
            for (int nf = 0; nf < 4; nf++) acc[1][nf] = wmma_f16(af1, bf[nf], acc[1][nf]);
        }
    }

#pragma unroll
    for (int mi = 0; mi < 2; mi++)
#pragma unroll
        for (int nf = 0; nf < 4; nf++) {
            int gn = n0 + wn * 64 + nf * 16 + r16;
            float bv = bias[gn];
#pragma unroll
            for (int r = 0; r < 8; r++) {
                int gm = m0 + wm * 32 + mi * 16 + r + 8 * hi;
                float val = acc[mi][nf][r] + bv;
                int bb = gm >> 10, t = gm & 1023;     // gm = b*T + t
                int h = gn >> 6,  d = gn & 63;        // gn = h*D + d
                if (MODE == 0) {
                    ((_Float16*)Out)[((size_t)(bb * H_ + h) * T_ + t) * D_ + d] = (_Float16)val;
                } else if (MODE == 1) {
                    ((_Float16*)Out)[((size_t)(bb * H_ + h) * D_ + d) * T_ + t] = (_Float16)val;
                } else {
                    ((float*)Out)[(size_t)gm * C_ + gn] = val;
                }
            }
        }
}

// ---------------- flash attention ----------------
// qh,kh: f16 [B,H,T,D]; vT: f16 [B,H,D,T]; mask: int32 [B,T]; out: f16 [B,T,C]
__global__ __launch_bounds__(256) void attn_kernel(const _Float16* __restrict__ qh,
                                                   const _Float16* __restrict__ kh,
                                                   const _Float16* __restrict__ vT,
                                                   const int* __restrict__ mask,
                                                   _Float16* __restrict__ outh) {
    constexpr int KSTR = 72;                 // halves: 144B pitch, aligned + conflict-free
    __shared__ _Float16 lK[2][64 * KSTR];
    __shared__ _Float16 lV[2][64 * KSTR];
    __shared__ _Float16 lP[8 * 16 * KSTR];   // wave-private P tiles

    const float NEGINF = -__builtin_inff();
    const int tid = threadIdx.x, lane = tid & 31, wv = tid >> 5;
    const int r16 = lane & 15, hi = lane >> 4;
    const int bh = blockIdx.x >> 3;          // 8 q-blocks of 128 rows each
    const int qb = blockIdx.x & 7;
    const int b  = bh >> 4;
    const int h  = bh & 15;
    const int q0 = qb * 128 + wv * 16;       // first q row of this wave

    const _Float16* qbase = qh + (size_t)bh * T_ * D_;
    const _Float16* kbase = kh + (size_t)bh * T_ * D_;
    const _Float16* vbase = vT + (size_t)bh * D_ * T_;
    const int* mrow = mask + b * T_;

    // Q fragments (16 rows x 64 d), pre-scaled by 1/sqrt(D) = 0.125 (exact in f16)
    half16 qf[2];
    {
        const _Float16* p = qbase + (size_t)(q0 + r16) * D_ + hi * 8;
        union { half16 v; half8 g[2]; } u0, u1;
        u0.g[0] = *(const half8*)(p);
        u0.g[1] = *(const half8*)(p + 16);
        u1.g[0] = *(const half8*)(p + 32);
        u1.g[1] = *(const half8*)(p + 48);
#pragma unroll
        for (int i = 0; i < 16; i++) { u0.v[i] *= (_Float16)0.125f; u1.v[i] *= (_Float16)0.125f; }
        qf[0] = u0.v; qf[1] = u1.v;
    }

    float m_r[8], l_r[8];
    float8 o[4];
#pragma unroll
    for (int r = 0; r < 8; r++) { m_r[r] = NEGINF; l_r[r] = 0.0f; }
#pragma unroll
    for (int df = 0; df < 4; df++)
#pragma unroll
        for (int i = 0; i < 8; i++) o[df][i] = 0.0f;

    const int srow = tid >> 2, sseg = tid & 3;
    _Float16* pw = &lP[wv * 16 * KSTR];

    auto issue_kv = [&](int kt, int buf) {
        const _Float16* gk = kbase + (size_t)(kt + srow) * D_ + sseg * 16;
        _Float16* dk = &lK[buf][srow * KSTR + sseg * 16];
        cp16(gk, dk); cp16(gk + 8, dk + 8);
        const _Float16* gv = vbase + (size_t)srow * T_ + kt + sseg * 16;
        _Float16* dv = &lV[buf][srow * KSTR + sseg * 16];
        cp16(gv, dv); cp16(gv + 8, dv + 8);
    };

    issue_kv(0, 0);
    int buf = 0;
    for (int kt = 0; kt < T_; kt += 64, buf ^= 1) {
        async_wait();
        __syncthreads();
        if (kt + 64 < T_) issue_kv(kt + 64, buf ^ 1);
        if (kt + 128 < T_) {
            __builtin_prefetch(kbase + (size_t)(kt + 128 + srow) * D_ + sseg * 16, 0, 1);
            __builtin_prefetch(vbase + (size_t)srow * T_ + kt + 128 + sseg * 16, 0, 1);
        }

        // S = Q K^T (scaled), 16 x 64 per wave
        float8 s[4];
#pragma unroll
        for (int nf = 0; nf < 4; nf++) {
            half16 b0 = ld_frag(lK[buf], nf * 16, KSTR, 0);
            half16 b1 = ld_frag(lK[buf], nf * 16, KSTR, 32);
            float8 z;
#pragma unroll
            for (int i = 0; i < 8; i++) z[i] = 0.0f;
            z = wmma_f16(qf[0], b0, z);
            z = wmma_f16(qf[1], b1, z);
            s[nf] = z;
        }
        // mask (per key column)
#pragma unroll
        for (int nf = 0; nf < 4; nf++) {
            int col = kt + nf * 16 + r16;
            if (mrow[col] == 0) {
#pragma unroll
                for (int i = 0; i < 8; i++) s[nf][i] = NEGINF;
            }
        }
        // online softmax: row max over 4 frags + 16 lanes (xor stays in 16-lane group)
        float tmax[8];
#pragma unroll
        for (int r = 0; r < 8; r++)
            tmax[r] = fmaxf(fmaxf(s[0][r], s[1][r]), fmaxf(s[2][r], s[3][r]));
#pragma unroll
        for (int r = 0; r < 8; r++) {
            tmax[r] = fmaxf(tmax[r], swz<1>(tmax[r]));
            tmax[r] = fmaxf(tmax[r], swz<2>(tmax[r]));
            tmax[r] = fmaxf(tmax[r], swz<4>(tmax[r]));
            tmax[r] = fmaxf(tmax[r], swz<8>(tmax[r]));
        }

        float corr[8], rsum[8];
#pragma unroll
        for (int r = 0; r < 8; r++) {
            float mn = fmaxf(m_r[r], tmax[r]);
            float msafe = (mn == NEGINF) ? 0.0f : mn;   // fully-masked guard
            corr[r] = __expf(m_r[r] - msafe);
            m_r[r] = mn;
            rsum[r] = 0.0f;
            int row = r + 8 * hi;
#pragma unroll
            for (int nf = 0; nf < 4; nf++) {
                float pv = __expf(s[nf][r] - msafe);
                rsum[r] += pv;
                pw[row * KSTR + nf * 16 + r16] = (_Float16)pv;
            }
        }
#pragma unroll
        for (int r = 0; r < 8; r++) {
            rsum[r] += swz<1>(rsum[r]);
            rsum[r] += swz<2>(rsum[r]);
            rsum[r] += swz<4>(rsum[r]);
            rsum[r] += swz<8>(rsum[r]);
        }
#pragma unroll
        for (int r = 0; r < 8; r++) l_r[r] = l_r[r] * corr[r] + rsum[r];
#pragma unroll
        for (int df = 0; df < 4; df++)
#pragma unroll
            for (int i = 0; i < 8; i++) o[df][i] *= corr[i];

        // O += P V  (P re-laid-out as A-frags via wave-private LDS)
        half16 pf0 = ld_frag(pw, 0, KSTR, 0);
        half16 pf1 = ld_frag(pw, 0, KSTR, 32);
#pragma unroll
        for (int df = 0; df < 4; df++) {
            half16 v0 = ld_frag(lV[buf], df * 16, KSTR, 0);
            half16 v1 = ld_frag(lV[buf], df * 16, KSTR, 32);
            o[df] = wmma_f16(pf0, v0, o[df]);
            o[df] = wmma_f16(pf1, v1, o[df]);
        }
    }

    float inv[8];
#pragma unroll
    for (int r = 0; r < 8; r++) inv[r] = 1.0f / l_r[r];
#pragma unroll
    for (int df = 0; df < 4; df++)
#pragma unroll
        for (int r = 0; r < 8; r++) {
            int t = q0 + r + 8 * hi;
            int d = df * 16 + r16;
            outh[((size_t)(b * T_ + t)) * C_ + h * D_ + d] = (_Float16)(o[df][r] * inv[r]);
        }
}

// ---------------- LayerNorm over last dim (1024) ----------------
__global__ __launch_bounds__(256) void ln_kernel(const float* __restrict__ y,
                                                 const float* __restrict__ g,
                                                 const float* __restrict__ bt,
                                                 float* __restrict__ out) {
    __shared__ float red[2][8];
    const int row = blockIdx.x;
    const int tid = threadIdx.x, lane = tid & 31, wv = tid >> 5;
    const float* yr = y + (size_t)row * C_;
    float4 v = ((const float4*)yr)[tid];
    float s  = v.x + v.y + v.z + v.w;
    float s2 = v.x * v.x + v.y * v.y + v.z * v.z + v.w * v.w;
    s += swz<1>(s);  s += swz<2>(s);  s += swz<4>(s);  s += swz<8>(s);  s += swz<16>(s);
    s2 += swz<1>(s2); s2 += swz<2>(s2); s2 += swz<4>(s2); s2 += swz<8>(s2); s2 += swz<16>(s2);
    if (lane == 0) { red[0][wv] = s; red[1][wv] = s2; }
    __syncthreads();
    float ts = 0.0f, ts2 = 0.0f;
#pragma unroll
    for (int w = 0; w < 8; w++) { ts += red[0][w]; ts2 += red[1][w]; }
    float mu = ts * (1.0f / C_);
    float var = ts2 * (1.0f / C_) - mu * mu;
    float rs = rsqrtf(var + 1e-5f);
    float4 gg = ((const float4*)g)[tid];
    float4 bb = ((const float4*)bt)[tid];
    float4 r;
    r.x = (v.x - mu) * rs * gg.x + bb.x;
    r.y = (v.y - mu) * rs * gg.y + bb.y;
    r.z = (v.z - mu) * rs * gg.z + bb.z;
    r.w = (v.w - mu) * rs * gg.w + bb.w;
    ((float4*)(out + (size_t)row * C_))[tid] = r;
}

extern "C" void kernel_launch(void* const* d_in, const int* in_sizes, int n_in,
                              void* d_out, int out_size, void* d_ws, size_t ws_size,
                              hipStream_t stream) {
    const float* x    = (const float*)d_in[0];
    const int*   mask = (const int*)d_in[1];
    const float* Wq = (const float*)d_in[2];  const float* bq = (const float*)d_in[3];
    const float* Wk = (const float*)d_in[4];  const float* bk = (const float*)d_in[5];
    const float* Wv = (const float*)d_in[6];  const float* bv = (const float*)d_in[7];
    const float* Wo = (const float*)d_in[8];  const float* bo = (const float*)d_in[9];
    const float* ln_g = (const float*)d_in[10];
    const float* ln_b = (const float*)d_in[11];
    float* out = (float*)d_out;

    char* w = (char*)d_ws;
    _Float16* xh    = (_Float16*)w;  w += (size_t)M_ * C_ * 2;      // 8 MB
    _Float16* WqT   = (_Float16*)w;  w += (size_t)C_ * C_ * 2;      // 2 MB
    _Float16* WkT   = (_Float16*)w;  w += (size_t)C_ * C_ * 2;
    _Float16* WvT   = (_Float16*)w;  w += (size_t)C_ * C_ * 2;
    _Float16* WoT   = (_Float16*)w;  w += (size_t)C_ * C_ * 2;
    _Float16* qhB   = (_Float16*)w;  w += (size_t)M_ * C_ * 2;      // [B,H,T,D]
    _Float16* khB   = (_Float16*)w;  w += (size_t)M_ * C_ * 2;      // [B,H,T,D]
    _Float16* vTB   = (_Float16*)w;  w += (size_t)M_ * C_ * 2;      // [B,H,D,T]
    _Float16* attnh = (_Float16*)w;  w += (size_t)M_ * C_ * 2;      // [B,T,C]
    float*    y     = (float*)w;     w += (size_t)M_ * C_ * 4;      // 16 MB

    const int n = M_ * C_;
    conv_f16_kernel<<<n / 4 / 256, 256, 0, stream>>>(x, xh, n);
    dim3 tgrid(C_ / 64, C_ / 64);
    transpose_f16_kernel<<<tgrid, 256, 0, stream>>>(Wq, WqT);
    transpose_f16_kernel<<<tgrid, 256, 0, stream>>>(Wk, WkT);
    transpose_f16_kernel<<<tgrid, 256, 0, stream>>>(Wv, WvT);
    transpose_f16_kernel<<<tgrid, 256, 0, stream>>>(Wo, WoT);

    dim3 ggrid(C_ / 128, M_ / 128);   // (8, 32)
    gemm_wmma_kernel<0><<<ggrid, 256, 0, stream>>>(xh, WqT, bq, (void*)qhB);
    gemm_wmma_kernel<0><<<ggrid, 256, 0, stream>>>(xh, WkT, bk, (void*)khB);
    gemm_wmma_kernel<1><<<ggrid, 256, 0, stream>>>(xh, WvT, bv, (void*)vTB);

    attn_kernel<<<B_ * H_ * (T_ / 128), 256, 0, stream>>>(qhB, khB, vTB, mask, attnh);

    gemm_wmma_kernel<2><<<ggrid, 256, 0, stream>>>(attnh, WoT, bo, (void*)y);

    ln_kernel<<<M_, 256, 0, stream>>>(y, ln_g, ln_b, out);
}
